// GNNModel_3264175145417
// MI455X (gfx1250) — compile-verified, matrix operand order checked by reading, not run on hardware
//
#include <hip/hip_runtime.h>
#include <hip/hip_bf16.h>

typedef __attribute__((ext_vector_type(16))) _Float16 v16h;
typedef __attribute__((ext_vector_type(8)))  _Float16 v8h;
typedef __attribute__((ext_vector_type(8)))  float    v8f;

#define N_NODES 100000
#define N_EDGES 1600000
#define IN_CH   128
#define HID     64

// ---------------- utility kernels ----------------

__global__ void fill_f32(float* __restrict__ p, float v, long long n) {
    long long i = (long long)blockIdx.x * blockDim.x + threadIdx.x;
    if (i < n) p[i] = v;
}

__global__ void cvt_f32_to_f16(const float* __restrict__ s,
                               _Float16* __restrict__ d, int n) {
    int i = blockIdx.x * blockDim.x + threadIdx.x;
    if (i < n) d[i] = (_Float16)s[i];
}

__global__ void degree_kernel(const int* __restrict__ ei,
                              float* __restrict__ deg, int E) {
    int e = blockIdx.x * blockDim.x + threadIdx.x;
    if (e < E) atomicAdd(&deg[ei[E + e]], 1.0f);
}

__global__ void rsqrt_kernel(float* __restrict__ dinv, int n) {
    int i = blockIdx.x * blockDim.x + threadIdx.x;
    if (i < n) dinv[i] = rsqrtf(dinv[i]);   // deg >= 1 always (self-loop)
}

__global__ void bias_relu_kernel(float* __restrict__ a,
                                 const float* __restrict__ b, long long n) {
    long long i = (long long)blockIdx.x * blockDim.x + threadIdx.x;
    if (i < n) {
        float v = a[i] + b[i & 63];
        a[i] = v > 0.0f ? v : 0.0f;
    }
}

__global__ void bias_add_kernel(float* __restrict__ a,
                                const float* __restrict__ b, long long n) {
    long long i = (long long)blockIdx.x * blockDim.x + threadIdx.x;
    if (i < n) a[i] = a[i] + b[i & 63];
}

// ---------------- WMMA GEMM: C[M,64] = A[M,K](f32) * Bt[64,K](f16)^T ----------------
// One wave per 16-row M-tile; 4 x v_wmma_f32_16x16x32_f16 per k-step cover N=64.
// A-fragment layout (ISA 16-bit A 16x32): lane%16 = M row; lane/16 selects K-half:
//   elems 0..7  <- K = k0 + (lane<16?0:8)  + 0..7   (contiguous in row-major A)
//   elems 8..15 <- K = k0 + 16 + (lane<16?0:8) + 0..7
// B-fragment mirrors this with lane%16 = N column; Bt is [64,K] row-major (= W as given),
// so each half of the fragment is one contiguous v8h (b128) load.
__global__ __launch_bounds__(128) void gemm_wmma_f16(
    const float*    __restrict__ A,
    const _Float16* __restrict__ Bt,
    float*          __restrict__ C,
    int M, int K)
{
    const int lane  = threadIdx.x & 31;
    const int mtile = (blockIdx.x << 2) + (threadIdx.x >> 5);
    const int mt_count = (M + 15) >> 4;
    if (mtile >= mt_count) return;            // uniform per wave

    const int row   = mtile * 16 + (lane & 15);
    const int rowc  = row < M ? row : M - 1;  // clamp (keeps EXEC all-1s for WMMA)
    const int khalf = (lane >> 4) << 3;       // 0 or 8
    const int col   = lane & 15;

    v8f acc[4] = {};
    const float* arow = A + (size_t)rowc * K;

    for (int k0 = 0; k0 < K; k0 += 32) {
        const int ka = k0 + khalf;
        const int kb = k0 + 16 + khalf;

        v16h afrag;
#pragma unroll
        for (int j = 0; j < 8; ++j) {
            afrag[j]     = (_Float16)arow[ka + j];
            afrag[8 + j] = (_Float16)arow[kb + j];
        }

#pragma unroll
        for (int t = 0; t < 4; ++t) {
            const _Float16* brow = Bt + (size_t)(t * 16 + col) * K;
            union { v16h v; v8h h[2]; } bu;
            bu.h[0] = *(const v8h*)(brow + ka);
            bu.h[1] = *(const v8h*)(brow + kb);
            acc[t] = __builtin_amdgcn_wmma_f32_16x16x32_f16(
                false, afrag, false, bu.v, (short)0, acc[t], false, false);
        }
    }

    // C/D layout: VGPR r, lanes 0-15 -> M = r, lanes 16-31 -> M = 8 + r; N = lane%16
    const int mbase = mtile * 16 + ((lane >> 4) << 3);
#pragma unroll
    for (int r = 0; r < 8; ++r) {
        const int m = mbase + r;
        if (m < M) {
#pragma unroll
            for (int t = 0; t < 4; ++t)
                C[(size_t)m * 64 + t * 16 + col] = acc[t][r];
        }
    }
}

// ---------------- edge scatter: out[dst] += dinv[src]*dinv[dst] * h[src] ----------------
// 32 lanes per edge, each lane owns a float2 of the 64 channels.
__global__ void scatter_kernel(const int* __restrict__ ei,
                               const float* __restrict__ h,
                               const float* __restrict__ dinv,
                               float* __restrict__ out,
                               int E, int N) {
    long long tid = (long long)blockIdx.x * blockDim.x + threadIdx.x;
    long long eid = tid >> 5;
    if (eid >= (long long)E + N) return;
    int c = (int)(tid & 31) * 2;

    int s, d;
    if (eid < E) { s = ei[eid]; d = ei[E + eid]; }
    else         { s = d = (int)(eid - E); }       // self-loop

    float norm = dinv[s] * dinv[d];
    float2 v = *(const float2*)(h + (size_t)s * 64 + c);
    float* op = out + (size_t)d * 64 + c;
    atomicAdd(op,     norm * v.x);
    atomicAdd(op + 1, norm * v.y);
}

// ---------------- launch ----------------

extern "C" void kernel_launch(void* const* d_in, const int* in_sizes, int n_in,
                              void* d_out, int out_size, void* d_ws, size_t ws_size,
                              hipStream_t stream) {
    (void)in_sizes; (void)n_in; (void)out_size; (void)ws_size;

    const float* x  = (const float*)d_in[0];
    const int*   ei = (const int*)d_in[1];
    const float* W1 = (const float*)d_in[2];
    const float* b1 = (const float*)d_in[3];
    const float* W2 = (const float*)d_in[4];
    const float* b2 = (const float*)d_in[5];
    float* out = (float*)d_out;

    char* ws = (char*)d_ws;
    auto al = [](size_t v) { return (v + 255) & ~(size_t)255; };
    size_t off = 0;
    float*    dinv = (float*)(ws + off);    off = al(off + (size_t)N_NODES * 4);
    _Float16* w1h  = (_Float16*)(ws + off); off = al(off + (size_t)HID * IN_CH * 2);
    _Float16* w2h  = (_Float16*)(ws + off); off = al(off + (size_t)HID * HID * 2);
    float*    hpre = (float*)(ws + off);    off = al(off + (size_t)N_NODES * HID * 4);
    float*    agg1 = (float*)(ws + off);    off = al(off + (size_t)N_NODES * HID * 4);

    const long long NF = (long long)N_NODES * HID;

    // init: deg=1 (self-loop), zero accumulators, convert weights to f16
    fill_f32<<<(N_NODES + 255) / 256, 256, 0, stream>>>(dinv, 1.0f, N_NODES);
    fill_f32<<<(unsigned)((NF + 255) / 256), 256, 0, stream>>>(agg1, 0.0f, NF);
    fill_f32<<<(unsigned)((NF + 255) / 256), 256, 0, stream>>>(out, 0.0f, NF);
    cvt_f32_to_f16<<<(HID * IN_CH + 255) / 256, 256, 0, stream>>>(W1, w1h, HID * IN_CH);
    cvt_f32_to_f16<<<(HID * HID + 255) / 256, 256, 0, stream>>>(W2, w2h, HID * HID);

    // normalization
    degree_kernel<<<(N_EDGES + 255) / 256, 256, 0, stream>>>(ei, dinv, N_EDGES);
    rsqrt_kernel<<<(N_NODES + 255) / 256, 256, 0, stream>>>(dinv, N_NODES);

    const int mt = (N_NODES + 15) / 16;                 // 6250 M-tiles
    const unsigned gemm_blocks = (mt + 3) / 4;          // 4 waves / block
    const long long sthreads = ((long long)N_EDGES + N_NODES) * 32;
    const unsigned sblocks = (unsigned)((sthreads + 255) / 256);

    // layer 1
    gemm_wmma_f16<<<gemm_blocks, 128, 0, stream>>>(x, w1h, hpre, N_NODES, IN_CH);
    scatter_kernel<<<sblocks, 256, 0, stream>>>(ei, hpre, dinv, agg1, N_EDGES, N_NODES);
    bias_relu_kernel<<<(unsigned)((NF + 255) / 256), 256, 0, stream>>>(agg1, b1, NF);

    // layer 2
    gemm_wmma_f16<<<gemm_blocks, 128, 0, stream>>>(agg1, w2h, hpre, N_NODES, HID);
    scatter_kernel<<<sblocks, 256, 0, stream>>>(ei, hpre, dinv, out, N_EDGES, N_NODES);
    bias_add_kernel<<<(unsigned)((NF + 255) / 256), 256, 0, stream>>>(out, b2, NF);
}
